// SparseConvBlock_3968549781704
// MI455X (gfx1250) — matrix operand order repaired and checked by reading, not compiled, not run on hardware
//
#include <hip/hip_runtime.h>

#define N_VOX   500000
#define K_OFF   27
#define CIN     32
#define COUT    32
#define N_TILES (N_VOX / 16)      // 31250, exact
#define BN_EPS  1e-5f

#define R1_BLOCKS 125             // 31250 tiles / 250 per block
#define TILES_PER_R1 250

typedef __attribute__((ext_vector_type(16))) __bf16 v16bf;
typedef __attribute__((ext_vector_type(8)))  __bf16 v8bf;
typedef __attribute__((ext_vector_type(8)))  float  v8f;

// ---- workspace layout (bytes) ----
// Wt:      27*32*32*2            =    55,296 B @ 0
// featsBf: (500000+1)*32*2       = 32,000,064 B @ 65,536   (row N_VOX is all-zero)
// partial: 31250*64*4            =  8,000,000 B @ 32,065,600
// sums:    64*4                  =       256 B @ 40,065,600
// partial2:125*64*4              =    32,000 B @ 40,065,856
#define WT_OFF     0u
#define FEATS_OFF  65536u
#define PART_OFF   (FEATS_OFF + 32000064u)
#define SUMS_OFF   (PART_OFF + 8000000u)
#define PART2_OFF  (SUMS_OFF + 256u)

// ---------------- prep: f32 -> bf16 (+ trailing zero row) ----------------
__global__ void cvt_feats_kernel(const float* __restrict__ f,
                                 unsigned short* __restrict__ fb) {
    long i = (long)blockIdx.x * blockDim.x + threadIdx.x;
    if (i < (long)(N_VOX + 1) * CIN) {
        float v = (i < (long)N_VOX * CIN) ? f[i] : 0.0f;
        ((__bf16*)fb)[i] = (__bf16)v;
    }
}

// weight[k][cin][cout] f32  ->  Wt[k][cout][cin] bf16 (B-fragment friendly)
__global__ void cvt_w_kernel(const float* __restrict__ w,
                             unsigned short* __restrict__ wt) {
    int i = blockIdx.x * blockDim.x + threadIdx.x;
    if (i < K_OFF * CIN * COUT) {
        int cin  = i & 31;
        int cout = (i >> 5) & 31;
        int k    = i >> 10;
        ((__bf16*)wt)[i] = (__bf16)w[(k * CIN + cin) * COUT + cout];
    }
}

// ---------------- main conv: gather + WMMA ----------------
__global__ __launch_bounds__(128)
void conv_wmma_kernel(const unsigned short* __restrict__ featsBf,  // [N+1][32] bf16
                      const unsigned short* __restrict__ Wt,       // [27][32][32] bf16 (k,cout,cin)
                      const int* __restrict__ nbr_idx,             // [N][27]
                      const int* __restrict__ nbr_mask,            // [N][27]
                      float* __restrict__ out,                     // [N][32] pre-BN
                      float* __restrict__ partial) {               // [N_TILES][64]
    const int wave = threadIdx.x >> 5;
    const int lane = threadIdx.x & 31;
    const int tile = blockIdx.x * 4 + wave;
    if (tile >= N_TILES) return;                 // wave-uniform exit; EXEC all-1 for WMMA

    const int  voxBase = tile * 16;
    const int  mrow    = lane & 15;              // A-matrix row M handled by this lane
    const int  half    = lane >> 4;              // half-wave id
    const int  aOff    = half ? 16 : 0;          // K-offset of first A chunk for this lane
    const int  ncol    = lane & 15;              // B/N column
    const int  cinBase = half ? 16 : 0;          // B K-range for this lane

    const long base = (long)(voxBase + mrow) * K_OFF;
    const int* nip = nbr_idx  + base;
    const int* nmp = nbr_mask + base;

    v8f c0 = {};   // Cout 0..15
    v8f c1 = {};   // Cout 16..31

    for (int k = 0; k < K_OFF; ++k) {
        const int idxRaw = nip[k];
        const int msk    = nmp[k];
        const int idx    = msk ? idxRaw : N_VOX;   // row N_VOX is all-zero -> masked gather

        // ---- A fragment: lane<16 gets K0-15; lane>=16 gets K16-31 (contiguous)
        const unsigned short* frow = featsBf + (long)idx * CIN;
        v8bf alo = *(const v8bf*)(frow + aOff);
        v8bf ahi = *(const v8bf*)(frow + aOff + 8);
        v16bf a;
        #pragma unroll
        for (int i = 0; i < 8; ++i) { a[i] = alo[i]; a[8 + i] = ahi[i]; }

        // ---- B fragments: lane holds K=cinBase..cinBase+15 of column ncol (contiguous)
        const unsigned short* wk  = Wt + k * (CIN * COUT);
        const unsigned short* b0p = wk + ncol * CIN + cinBase;          // couts 0..15
        const unsigned short* b1p = wk + (16 + ncol) * CIN + cinBase;   // couts 16..31
        v8bf b0lo = *(const v8bf*)(b0p);
        v8bf b0hi = *(const v8bf*)(b0p + 8);
        v8bf b1lo = *(const v8bf*)(b1p);
        v8bf b1hi = *(const v8bf*)(b1p + 8);
        v16bf b0, b1;
        #pragma unroll
        for (int i = 0; i < 8; ++i) {
            b0[i] = b0lo[i]; b0[8 + i] = b0hi[i];
            b1[i] = b1lo[i]; b1[8 + i] = b1hi[i];
        }

        // D = A*B + C, f32 accumulate
        c0 = __builtin_amdgcn_wmma_f32_16x16x32_bf16(false, a, false, b0,
                                                     (short)0, c0, false, false);
        c1 = __builtin_amdgcn_wmma_f32_16x16x32_bf16(false, a, false, b1,
                                                     (short)0, c1, false, false);
    }

    // ---- store pre-BN conv output (C/D layout: vgpr j -> M = half*8 + j, N = ncol)
    const int ms = half * 8;
    #pragma unroll
    for (int j = 0; j < 8; ++j) {
        long o = (long)(voxBase + ms + j) * COUT + ncol;
        out[o]      = c0[j];
        out[o + 16] = c1[j];
    }

    // ---- per-channel partial sum / sumsq over this tile's 16 voxels
    float s0 = 0.f, q0 = 0.f, s1 = 0.f, q1 = 0.f;
    #pragma unroll
    for (int j = 0; j < 8; ++j) {
        s0 += c0[j]; q0 += c0[j] * c0[j];
        s1 += c1[j]; q1 += c1[j] * c1[j];
    }
    s0 += __shfl_down(s0, 16, 32); q0 += __shfl_down(q0, 16, 32);
    s1 += __shfl_down(s1, 16, 32); q1 += __shfl_down(q1, 16, 32);
    if (half == 0) {
        float* p = partial + (long)tile * 64;
        p[ncol]      = s0;   // sum,  ch 0..15
        p[16 + ncol] = s1;   // sum,  ch 16..31
        p[32 + ncol] = q0;   // sumsq,ch 0..15
        p[48 + ncol] = q1;   // sumsq,ch 16..31
    }
}

// ---------------- deterministic tree reduction, stage 1 ----------------
// 125 blocks x 256 threads: each block folds 250 tiles -> partial2[block][64]
__global__ __launch_bounds__(256)
void reduce1_kernel(const float* __restrict__ partial,
                    float* __restrict__ partial2) {
    __shared__ float sdata[256];
    const int t   = threadIdx.x;
    const int ch  = t & 63;
    const int sub = t >> 6;                       // 0..3
    const long tileBase = (long)blockIdx.x * TILES_PER_R1;
    float acc = 0.f;
    for (int j = sub; j < TILES_PER_R1; j += 4)
        acc += partial[(tileBase + j) * 64 + ch];
    sdata[t] = acc;
    __syncthreads();
    if (sub == 0)
        partial2[(long)blockIdx.x * 64 + ch] =
            sdata[ch] + sdata[64 + ch] + sdata[128 + ch] + sdata[192 + ch];
}

// ---------------- deterministic tree reduction, stage 2 ----------------
__global__ void reduce2_kernel(const float* __restrict__ partial2,
                               float* __restrict__ sums) {
    int t = threadIdx.x;                          // 0..63
    float acc = 0.f;
    #pragma unroll 5
    for (int b = 0; b < R1_BLOCKS; ++b) acc += partial2[b * 64 + t];
    sums[t] = acc;                                // [0..31]=sum(ch), [32..63]=sumsq(ch)
}

// ---------------- BN (batch stats) + ReLU, in place on d_out ----------------
__global__ void bn_relu_kernel(float* __restrict__ out,
                               const float* __restrict__ sums,
                               const float* __restrict__ gamma,
                               const float* __restrict__ beta) {
    long e = (long)blockIdx.x * blockDim.x + threadIdx.x;
    if (e >= (long)N_VOX * COUT) return;
    int c = (int)(e & 31);
    const float inv_n = 1.0f / (float)N_VOX;
    float mean  = sums[c] * inv_n;
    float var   = sums[32 + c] * inv_n - mean * mean;   // biased, matches jnp.var
    float scale = gamma[c] * rsqrtf(var + BN_EPS);
    float shift = beta[c] - mean * scale;
    float y = out[e] * scale + shift;
    out[e] = y > 0.f ? y : 0.f;
}

extern "C" void kernel_launch(void* const* d_in, const int* in_sizes, int n_in,
                              void* d_out, int out_size, void* d_ws, size_t ws_size,
                              hipStream_t stream) {
    const float* feats    = (const float*)d_in[0];
    const float* weight   = (const float*)d_in[1];
    const float* gamma    = (const float*)d_in[2];
    const float* beta     = (const float*)d_in[3];
    const int*   nbr_idx  = (const int*)d_in[4];
    const int*   nbr_mask = (const int*)d_in[5];
    float* out = (float*)d_out;

    char* ws = (char*)d_ws;
    unsigned short* Wt       = (unsigned short*)(ws + WT_OFF);
    unsigned short* featsBf  = (unsigned short*)(ws + FEATS_OFF);
    float*          partial  = (float*)(ws + PART_OFF);
    float*          sums     = (float*)(ws + SUMS_OFF);
    float*          partial2 = (float*)(ws + PART2_OFF);

    // 1) feats f32 -> bf16, plus one all-zero row at index N_VOX (16,000,032 elems)
    cvt_feats_kernel<<<62501, 256, 0, stream>>>(feats, featsBf);
    // 2) weights f32 [k][cin][cout] -> bf16 [k][cout][cin]
    cvt_w_kernel<<<(K_OFF * CIN * COUT + 255) / 256, 256, 0, stream>>>(weight, Wt);
    // 3) gather + WMMA conv (31250 tiles, 4 waves/block)
    conv_wmma_kernel<<<(N_TILES + 3) / 4, 128, 0, stream>>>(
        featsBf, Wt, nbr_idx, nbr_mask, out, partial);
    // 4) deterministic two-stage tree reduction of per-tile partials
    reduce1_kernel<<<R1_BLOCKS, 256, 0, stream>>>(partial, partial2);
    reduce2_kernel<<<1, 64, 0, stream>>>(partial2, sums);
    // 5) BN + ReLU in place
    bn_relu_kernel<<<62500, 256, 0, stream>>>(out, sums, gamma, beta);
}
